// GraphSAGELinkPredictor_64888365907988
// MI455X (gfx1250) — matrix operand order repaired and checked by reading, compiled, not run on hardware
//
#include <hip/hip_runtime.h>
#include <hip/hip_bf16.h>
#include <stdint.h>

typedef __attribute__((ext_vector_type(16))) _Float16 v16h;
typedef __attribute__((ext_vector_type(8)))  _Float16 v8h;
typedef __attribute__((ext_vector_type(8)))  float    v8f;

#define NN   100000
#define HIDD 128
#define NE   1600000
#define NP   500000

static __device__ __forceinline__ v16h cat16(v8h lo, v8h hi) {
  v16h r;
#pragma unroll
  for (int i = 0; i < 8; ++i) { r[i] = lo[i]; r[i + 8] = hi[i]; }
  return r;
}

// A-operand chunk (16x32 f16 tile of a row-major [., 128] matrix).
// lane l: M = l%16; halfs[0..7] = K kb+(hi?8:0).., halfs[8..15] = K kb+16+(hi?8:0)..
static __device__ __forceinline__ v16h load_a(const _Float16* base, int row, int kb, bool hi) {
  const _Float16* p = base + row * HIDD + kb + (hi ? 8 : 0);
  v8h lo = *(const v8h*)(p);
  v8h h8 = *(const v8h*)(p + 16);
  return cat16(lo, h8);
}

// B-operand chunk (32x16 f16). lane l: N = l%16 -> weight row j, halfs[0..7] = K kb+(hi?8:0).., halfs[8..15] = K kb+16+(hi?8:0)..
static __device__ __forceinline__ v16h load_b(const _Float16* w, int j, int kb, bool hi) {
  const _Float16* p = w + j * HIDD + kb + (hi ? 8 : 0);
  v8h lo = *(const v8h*)(p);
  v8h h8 = *(const v8h*)(p + 16);
  return cat16(lo, h8);
}

// CDNA5 packed-f16 atomic add (GLOBAL_ATOMIC_PK_ADD_F16, no return -> STOREcnt).
static __device__ __forceinline__ void pk_atomic_fadd(_Float16* p, uint32_t two_halfs) {
  asm volatile("global_atomic_pk_add_f16 %0, %1, off"
               :: "v"((uint64_t)(uintptr_t)p), "v"(two_halfs) : "memory");
}

__global__ void k_f32_to_f16(const float* __restrict__ in, _Float16* __restrict__ out, int n) {
  int i = blockIdx.x * blockDim.x + threadIdx.x;
  if (i < n) out[i] = (_Float16)in[i];
}

__global__ void k_zero(float* p, int n) {
  int i = blockIdx.x * blockDim.x + threadIdx.x;
  if (i < n) p[i] = 0.0f;
}

// One edge per 32 lanes; each lane accumulates 4 dims as two packed-f16 atomics.
// msg buffer (25.6 MB f16) is L2-resident on MI455X (192 MB L2): atomics resolve at L2.
__global__ void k_scatter(const _Float16* __restrict__ feat, const int* __restrict__ src,
                          const int* __restrict__ dst, _Float16* __restrict__ msg,
                          float* __restrict__ cnt, int n_edges) {
  int t = blockIdx.x * blockDim.x + threadIdx.x;
  int e = t >> 5;
  int c = t & 31;
  if (e >= n_edges) return;
  int s = src[e], d = dst[e];
  const uint32_t* f = (const uint32_t*)(feat + s * HIDD + c * 4);
  _Float16* m = msg + d * HIDD + c * 4;
  uint32_t f0 = f[0];
  uint32_t f1 = f[1];
  pk_atomic_fadd(m, f0);
  pk_atomic_fadd(m + 2, f1);
  if (c == 0) atomicAdd(cnt + d, 1.0f);
}

// out = act((msg/cnt) @ Wl^T + bias + root @ Wr^T).
// One wave per 16-row stripe computes all 8 column tiles: activations read ONCE,
// mean-divide fused into the A fragment (scale is constant per lane-row).
__global__ void __launch_bounds__(64) k_sage_gemm(
    const _Float16* __restrict__ msg, const float* __restrict__ cnt,
    const _Float16* __restrict__ root,
    const _Float16* __restrict__ Wl, const _Float16* __restrict__ Wr,
    const float* __restrict__ bias, _Float16* __restrict__ out, int relu) {
  int wave = (blockIdx.x * blockDim.x + threadIdx.x) >> 5;  // 0..6249 row stripes
  int lane = threadIdx.x & 31;
  int m = lane & 15;
  bool hi = lane >= 16;
  int row = wave * 16 + m;
  _Float16 scale = (_Float16)(1.0f / fmaxf(cnt[row], 1.0f));

  v8f acc[8];
#pragma unroll
  for (int jt = 0; jt < 8; ++jt) acc[jt] = (v8f){};

#pragma unroll
  for (int kbi = 0; kbi < 4; ++kbi) {
    int kb = kbi * 32;
    v16h a1 = load_a(msg, row, kb, hi);
#pragma unroll
    for (int q = 0; q < 16; ++q) a1[q] *= scale;      // fused mean aggregation
    v16h a2 = load_a(root, row, kb, hi);
#pragma unroll
    for (int jt = 0; jt < 8; ++jt) {
      int j = jt * 16 + m;
      v16h b1 = load_b(Wl, j, kb, hi);
      acc[jt] = __builtin_amdgcn_wmma_f32_16x16x32_f16(false, a1, false, b1, (short)0, acc[jt], false, false);
      v16h b2 = load_b(Wr, j, kb, hi);
      acc[jt] = __builtin_amdgcn_wmma_f32_16x16x32_f16(false, a2, false, b2, (short)0, acc[jt], false, false);
    }
  }

#pragma unroll
  for (int jt = 0; jt < 8; ++jt) {
    float bv = bias[jt * 16 + m];
#pragma unroll
    for (int v = 0; v < 8; ++v) {
      float o = acc[jt][v] + bv;
      if (relu) o = fmaxf(o, 0.0f);
      int rr = wave * 16 + v + (hi ? 8 : 0);
      out[rr * HIDD + jt * 16 + m] = (_Float16)o;
    }
  }
}

// Full decoder per wave: 16 pairs -> z_i*z_j -> 128->128 -> 128->64 -> 64->1.
// Intermediates bounce through a wave-private LDS tile; wave-local ordering via s_wait_dscnt.
__global__ void __launch_bounds__(64) k_decoder(
    const _Float16* __restrict__ z, const int* __restrict__ pairs,
    const _Float16* __restrict__ Wm1, const float* __restrict__ bm1,
    const _Float16* __restrict__ Wm2, const float* __restrict__ bm2,
    const float* __restrict__ Wm3, const float* __restrict__ bm3,
    float* __restrict__ out) {
  __shared__ __align__(16) _Float16 hp1s[2][16 * HIDD];
  __shared__ __align__(16) float    hp2s[2][16 * 64];
  int w = threadIdx.x >> 5;
  int lane = threadIdx.x & 31;
  int m = lane & 15;
  bool hi = lane >= 16;
  int pairbase = (blockIdx.x * 2 + w) * 16;
  int pi = pairs[2 * (pairbase + m) + 0];
  int pj = pairs[2 * (pairbase + m) + 1];

  // p = z[i] ⊙ z[j], built directly in WMMA A layout (4 K-chunks of 32)
  v16h pa[4];
#pragma unroll
  for (int kbi = 0; kbi < 4; ++kbi) {
    v16h zi = load_a(z, pi, kbi * 32, hi);
    v16h zj = load_a(z, pj, kbi * 32, hi);
    v16h p;
#pragma unroll
    for (int q = 0; q < 16; ++q) p[q] = (_Float16)((float)zi[q] * (float)zj[q]);
    pa[kbi] = p;
  }

  // layer 1: [16x128] @ [128x128]^T, relu, spill f16 tile to LDS
  _Float16* h1 = &hp1s[w][0];
#pragma unroll
  for (int jt = 0; jt < 8; ++jt) {
    int j = jt * 16 + m;
    v8f acc = {};
#pragma unroll
    for (int kbi = 0; kbi < 4; ++kbi) {
      v16h b = load_b(Wm1, j, kbi * 32, hi);
      acc = __builtin_amdgcn_wmma_f32_16x16x32_f16(false, pa[kbi], false, b, (short)0, acc, false, false);
    }
    float bv = bm1[j];
#pragma unroll
    for (int v = 0; v < 8; ++v)
      h1[(v + (hi ? 8 : 0)) * HIDD + j] = (_Float16)fmaxf(acc[v] + bv, 0.0f);
  }
  asm volatile("s_wait_dscnt 0x0" ::: "memory");

  // layer 2: [16x128] @ [64x128]^T, relu, spill f32 tile to LDS
  v16h pa2[4];
#pragma unroll
  for (int kbi = 0; kbi < 4; ++kbi) pa2[kbi] = load_a(h1, m, kbi * 32, hi);
  float* h2 = &hp2s[w][0];
#pragma unroll
  for (int jt = 0; jt < 4; ++jt) {
    int j = jt * 16 + m;
    v8f acc = {};
#pragma unroll
    for (int kbi = 0; kbi < 4; ++kbi) {
      v16h b = load_b(Wm2, j, kbi * 32, hi);
      acc = __builtin_amdgcn_wmma_f32_16x16x32_f16(false, pa2[kbi], false, b, (short)0, acc, false, false);
    }
    float bv = bm2[j];
#pragma unroll
    for (int v = 0; v < 8; ++v)
      h2[(v + (hi ? 8 : 0)) * 64 + j] = fmaxf(acc[v] + bv, 0.0f);
  }
  asm volatile("s_wait_dscnt 0x0" ::: "memory");

  // layer 3: [16x64] @ [64] + b, one lane per pair
  if (lane < 16) {
    float acc = bm3[0];
#pragma unroll 8
    for (int k = 0; k < 64; ++k) acc += h2[lane * 64 + k] * Wm3[k];
    out[pairbase + lane] = acc;
  }
}

static inline int cdiv(int a, int b) { return (a + b - 1) / b; }

extern "C" void kernel_launch(void* const* d_in, const int* in_sizes, int n_in,
                              void* d_out, int out_size, void* d_ws, size_t ws_size,
                              hipStream_t stream) {
  (void)in_sizes; (void)n_in; (void)out_size; (void)ws_size;
  const float* x   = (const float*)d_in[0];
  const int*   ei  = (const int*)d_in[1];
  const int*   prs = (const int*)d_in[2];
  const float* W1l = (const float*)d_in[3];
  const float* b1l = (const float*)d_in[4];
  const float* W1r = (const float*)d_in[5];
  const float* W2l = (const float*)d_in[6];
  const float* b2l = (const float*)d_in[7];
  const float* W2r = (const float*)d_in[8];
  const float* Wm1 = (const float*)d_in[9];
  const float* bm1 = (const float*)d_in[10];
  const float* Wm2 = (const float*)d_in[11];
  const float* bm2 = (const float*)d_in[12];
  const float* Wm3 = (const float*)d_in[13];
  const float* bm3 = (const float*)d_in[14];
  float* out = (float*)d_out;
  const int* src = ei;
  const int* dst = ei + NE;

  // workspace carve-out (~103 MB), 256B aligned slices
  char* ws = (char*)d_ws;
  size_t off = 0;
  auto carve = [&](size_t bytes) {
    char* p = ws + off;
    off = (off + bytes + 255) & ~(size_t)255;
    return p;
  };
  _Float16* msg  = (_Float16*)carve((size_t)NN * HIDD * 2 + (size_t)NN * 4); // msg + cnt contiguous
  float*    cnt  = (float*)(msg + (size_t)NN * HIDD);
  _Float16* xh   = (_Float16*)carve((size_t)NN * HIDD * 2);
  _Float16* hh   = (_Float16*)carve((size_t)NN * HIDD * 2);
  _Float16* zh   = (_Float16*)carve((size_t)NN * HIDD * 2);
  _Float16* W1lh = (_Float16*)carve(HIDD * HIDD * 2);
  _Float16* W1rh = (_Float16*)carve(HIDD * HIDD * 2);
  _Float16* W2lh = (_Float16*)carve(HIDD * HIDD * 2);
  _Float16* W2rh = (_Float16*)carve(HIDD * HIDD * 2);
  _Float16* Wm1h = (_Float16*)carve(HIDD * HIDD * 2);
  _Float16* Wm2h = (_Float16*)carve(64 * HIDD * 2);

  const int B = 256;
  const int nZero = (NN * HIDD * 2 + NN * 4) / 4;  // msg (f16) + cnt (f32) as dwords

  // stage weights + input activations as f16
  k_f32_to_f16<<<cdiv(HIDD * HIDD, B), B, 0, stream>>>(W1l, W1lh, HIDD * HIDD);
  k_f32_to_f16<<<cdiv(HIDD * HIDD, B), B, 0, stream>>>(W1r, W1rh, HIDD * HIDD);
  k_f32_to_f16<<<cdiv(HIDD * HIDD, B), B, 0, stream>>>(W2l, W2lh, HIDD * HIDD);
  k_f32_to_f16<<<cdiv(HIDD * HIDD, B), B, 0, stream>>>(W2r, W2rh, HIDD * HIDD);
  k_f32_to_f16<<<cdiv(HIDD * HIDD, B), B, 0, stream>>>(Wm1, Wm1h, HIDD * HIDD);
  k_f32_to_f16<<<cdiv(64 * HIDD, B), B, 0, stream>>>(Wm2, Wm2h, 64 * HIDD);
  k_f32_to_f16<<<cdiv(NN * HIDD, B), B, 0, stream>>>(x, xh, NN * HIDD);

  // ---- layer 1 ----
  k_zero<<<cdiv(nZero, B), B, 0, stream>>>((float*)msg, nZero);
  k_scatter<<<cdiv(NE * 32, B), B, 0, stream>>>(xh, src, dst, msg, cnt, NE);
  k_sage_gemm<<<NN / 32, 64, 0, stream>>>(msg, cnt, xh, W1lh, W1rh, b1l, hh, 1);

  // ---- layer 2 ----
  k_zero<<<cdiv(nZero, B), B, 0, stream>>>((float*)msg, nZero);
  k_scatter<<<cdiv(NE * 32, B), B, 0, stream>>>(hh, src, dst, msg, cnt, NE);
  k_sage_gemm<<<NN / 32, 64, 0, stream>>>(msg, cnt, hh, W2lh, W2rh, b2l, zh, 0);

  // ---- decoder ----
  k_decoder<<<NP / 32, 64, 0, stream>>>(zh, prs, Wm1h, bm1, Wm2h, bm2, Wm3, bm3, out);
}
